// Patch5Model_15333033247106
// MI455X (gfx1250) — compile-verified, matrix operand order checked by reading, not run on hardware
//
#include <hip/hip_runtime.h>
#include <hip/hip_bf16.h>

// ---------------------------------------------------------------------------
// MI455X (gfx1250) implementation of Patch5Model.
// Dominant cost: backbone conv stem as implicit GEMM (M=10976,K=3072,N=2048)
// -> v_wmma_f32_16x16x32_f16.  Conv GEMM: 32x256 block tile, 8 wave32s in a
// 2x4 grid, 4 accumulators per wave (A-fragment reused 4x), LDS double
// buffering with ONE barrier per K-step.  Staging values are plain scalars
// (no arrays / no lambda captures) so they stay in VGPRs -- no scratch
// traffic in the hot loop.  Fused relu + global_atomic_add_f32 GAP epilogue.
// ---------------------------------------------------------------------------

typedef _Float16 h16;
typedef __attribute__((ext_vector_type(16))) _Float16 v16h;
typedef __attribute__((ext_vector_type(8)))  _Float16 v8h;
typedef __attribute__((ext_vector_type(8)))  float    v8f;

#define BATCH   32
#define HIMG    448
#define CROPSZ  224
#define CRES    2048
#define DMODEL  128
#define NPROP   6
#define KCONV   3072      // 3*32*32
#define FMW     7
#define FMP     49        // 7*7
#define NSTEP   (KCONV / 32)

__device__ __forceinline__ int imin(int a, int b) { return a < b ? a : b; }
__device__ __forceinline__ int imax(int a, int b) { return a > b ? a : b; }

// ---------------------------------------------------------------------------
__global__ void k_cvt(const float* __restrict__ in, h16* __restrict__ out, int n) {
    int i = blockIdx.x * 256 + threadIdx.x;
    if (i < n) out[i] = (h16)in[i];
}

// ---------------------------------------------------------------------------
// Conv stem as implicit GEMM.  A: images [nimg,3,224,224] f16 (im2col on the
// fly: row m = img*49 + (oy*7+ox), k = c*1024 + yy*32 + x -> 32 contiguous
// pixels per K-step).  B: conv_w [2048, 3072] f16 (N x K row-major).
// Iteration: global-load tile s+1 (scalars) -> fragments + 4 WMMA on tile s
// (covers memory latency) -> ds_store tile s+1 -> one barrier.
// ---------------------------------------------------------------------------
__global__ __launch_bounds__(256, 1)
void k_conv_gemm(const h16* __restrict__ img, const h16* __restrict__ W,
                 float* __restrict__ fm, float* __restrict__ gap) {
    __shared__ h16 As[2][32][40];    // 80B rows: 16B aligned, bank-friendly
    __shared__ h16 Bs[2][256][40];

    const int tid  = threadIdx.x;
    const int m0   = blockIdx.x * 32;
    const int n0   = blockIdx.y * 256;
    const int wave = tid >> 5, lane = tid & 31;
    const int wm   = wave & 1, wn = wave >> 1;      // 2 x 4 wave tile grid

    v8f acc[4] = {};

    // A-load slot: row ai (0..31), 8B chunk as (0..7)
    const int ai = tid >> 3, as = tid & 7;
    const int m     = m0 + ai;
    const int n_img = m / FMP;
    const int p     = m % FMP;
    const int oy    = p / FMW, ox = p % FMW;
    const size_t imgbase = (size_t)n_img * 3u * CROPSZ * CROPSZ
                         + (size_t)(oy * 32) * CROPSZ + (size_t)(ox * 32);

    // B-load slot: one weight row per thread, 4 x 16B chunks
    const size_t brow = (size_t)(n0 + tid) * KCONV;

    // scalar staging values: always promoted to VGPRs
    uint2 aReg;
    uint4 b0, b1, b2, b3;

    // prologue: tile 0 (kb = 0 -> c = 0, yy = 0) into buffer 0
    aReg = *(const uint2*)&img[imgbase + as * 4];
    b0 = *(const uint4*)&W[brow + 0];
    b1 = *(const uint4*)&W[brow + 8];
    b2 = *(const uint4*)&W[brow + 16];
    b3 = *(const uint4*)&W[brow + 24];
    *(uint2*)&As[0][ai][as * 4] = aReg;
    *(uint4*)&Bs[0][tid][0]     = b0;
    *(uint4*)&Bs[0][tid][8]     = b1;
    *(uint4*)&Bs[0][tid][16]    = b2;
    *(uint4*)&Bs[0][tid][24]    = b3;
    __syncthreads();

    for (int s = 0; s < NSTEP; ++s) {
        const int cur   = s & 1;
        const int nxt   = cur ^ 1;
        const bool more = (s + 1 < NSTEP);

        // issue global loads for tile s+1; latency hidden under the WMMAs
        if (more) {
            const int kb = (s + 1) * 32;
            const int c  = kb >> 10;                 // channel
            const int yy = (kb & 1023) >> 5;         // row within 32x32 patch
            const size_t aoff = imgbase + ((size_t)c * CROPSZ + (size_t)yy) * CROPSZ;
            aReg = *(const uint2*)&img[aoff + as * 4];
            b0 = *(const uint4*)&W[brow + kb + 0];
            b1 = *(const uint4*)&W[brow + kb + 8];
            b2 = *(const uint4*)&W[brow + kb + 16];
            b3 = *(const uint4*)&W[brow + kb + 24];
        }
        if (s + 2 < NSTEP) __builtin_prefetch(&W[brow + (s + 2) * 32], 0, 3);

        // fragments per CDNA5 16-bit A 16x32 layout (lanes<16: K0-7,K16-23;
        // lanes>=16: K8-15,K24-31).  B mirrored (weights stored N x K).
        const int mr = wm * 16 + (lane & 15);
        const int ko = (lane < 16) ? 0 : 8;
        v8h a_lo = *(const v8h*)&As[cur][mr][ko];
        v8h a_hi = *(const v8h*)&As[cur][mr][ko + 16];
        v16h av;
#pragma unroll
        for (int t = 0; t < 8; ++t) { av[t] = a_lo[t]; av[t + 8] = a_hi[t]; }

#pragma unroll
        for (int j = 0; j < 4; ++j) {
            const int nr = wn * 16 + j * 64 + (lane & 15);
            v8h b_lo = *(const v8h*)&Bs[cur][nr][ko];
            v8h b_hi = *(const v8h*)&Bs[cur][nr][ko + 16];
            v16h bv;
#pragma unroll
            for (int t = 0; t < 8; ++t) { bv[t] = b_lo[t]; bv[t + 8] = b_hi[t]; }
            acc[j] = __builtin_amdgcn_wmma_f32_16x16x32_f16(false, av, false, bv,
                                                            (short)0, acc[j],
                                                            false, false);
        }

        // stage tile s+1 (staging regs die here; not live across the barrier)
        if (more) {
            *(uint2*)&As[nxt][ai][as * 4] = aReg;
            *(uint4*)&Bs[nxt][tid][0]     = b0;
            *(uint4*)&Bs[nxt][tid][8]     = b1;
            *(uint4*)&Bs[nxt][tid][16]    = b2;
            *(uint4*)&Bs[nxt][tid][24]    = b3;
        }
        __syncthreads();
    }

    // Epilogue: C/D layout -> lanes 0-15: M=r, lanes 16-31: M=r+8; N=lane%16.
#pragma unroll
    for (int j = 0; j < 4; ++j) {
        const int ncol = n0 + wn * 16 + j * 64 + (lane & 15);
#pragma unroll
        for (int r = 0; r < 8; ++r) {
            const int mrow = m0 + wm * 16 + r + ((lane < 16) ? 0 : 8);
            float v = acc[j][r];
            v = v > 0.f ? v : 0.f;                     // relu
            const int nb = mrow / FMP, pp = mrow % FMP;
            if (fm) fm[((size_t)nb * CRES + ncol) * FMP + pp] = v;
            atomicAdd(&gap[(size_t)nb * CRES + ncol], v * (1.f / 49.f));
        }
    }
}

// ---------------------------------------------------------------------------
// Generic dense GEMM: C = relu?(A @ W^T + bias).  A [M,K] f16 rm, W [N,K] f16
// rm.  M % 32 == 0, N % 64 == 0, K % 32 == 0.  Optional f16 mirror store.
// ---------------------------------------------------------------------------
__global__ __launch_bounds__(256, 1)
void k_gemm(const h16* __restrict__ A, const h16* __restrict__ W,
            const float* __restrict__ bias, float* __restrict__ C,
            h16* __restrict__ Ch, int M, int N, int K, int relu) {
    __shared__ h16 As[32][40];
    __shared__ h16 Bs[64][40];

    const int tid  = threadIdx.x;
    const int m0   = blockIdx.x * 32;
    const int n0   = blockIdx.y * 64;
    const int wave = tid >> 5, lane = tid & 31;
    const int wm   = wave & 1, wn = wave >> 1;

    v8f acc = {};

    const int ai = tid >> 3, as = tid & 7;
    const int bj = tid >> 2, bs = tid & 3;
    const size_t arow = (size_t)(m0 + ai) * K;
    const size_t brow = (size_t)(n0 + bj) * K;

    for (int kb = 0; kb < K; kb += 32) {
        *(uint2*)&As[ai][as * 4] = *(const uint2*)&A[arow + kb + as * 4];
        *(uint4*)&Bs[bj][bs * 8] = *(const uint4*)&W[brow + kb + bs * 8];
        __syncthreads();

        const int mr = wm * 16 + (lane & 15);
        const int nr = wn * 16 + (lane & 15);
        const int ko = (lane < 16) ? 0 : 8;
        v8h a_lo = *(const v8h*)&As[mr][ko];
        v8h a_hi = *(const v8h*)&As[mr][ko + 16];
        v8h b_lo = *(const v8h*)&Bs[nr][ko];
        v8h b_hi = *(const v8h*)&Bs[nr][ko + 16];
        v16h av, bv;
#pragma unroll
        for (int t = 0; t < 8; ++t) {
            av[t] = a_lo[t]; av[t + 8] = a_hi[t];
            bv[t] = b_lo[t]; bv[t + 8] = b_hi[t];
        }
        acc = __builtin_amdgcn_wmma_f32_16x16x32_f16(false, av, false, bv,
                                                     (short)0, acc, false, false);
        __syncthreads();
    }

    const int ncol = n0 + wn * 16 + (lane & 15);
    const float bv0 = bias ? bias[ncol] : 0.f;
#pragma unroll
    for (int r = 0; r < 8; ++r) {
        const int mrow = m0 + wm * 16 + r + ((lane < 16) ? 0 : 8);
        float v = acc[r] + bv0;
        if (relu) v = v > 0.f ? v : 0.f;
        C[(size_t)mrow * N + ncol] = v;
        if (Ch) Ch[(size_t)mrow * N + ncol] = (h16)v;
    }
}

// ---------------------------------------------------------------------------
// COOI.get_coordinates: fm channel-sum -> avg-pool scores -> iterative
// argmax + 3x3 max-pool NMS erase -> 6 crop top-lefts per image.
// ---------------------------------------------------------------------------
__global__ void k_coords(const float* __restrict__ fm, const float* __restrict__ scale,
                         int* __restrict__ tl) {
    const int b = blockIdx.x;
    __shared__ float fsum[FMP];
    const int t = threadIdx.x;
    if (t < FMP) {
        float s = 0.f;
        const float* base = fm + (size_t)b * CRES * FMP + t;
        for (int o = 0; o < CRES; ++o) s += base[(size_t)o * FMP];
        fsum[t] = s;
    }
    __syncthreads();
    if (t != 0) return;

    const int sy = (int)scale[b * 2 + 0], sx = (int)scale[b * 2 + 1];
    const int smin = imin(sy, sx);
    const int by = (sy - smin) / 2, bx = (sx - smin) / 2;

    const int fkh[2] = {3, 2}, fkw[2] = {3, 2};
    const int nfil[2] = {3, 3};
    const int psz[2] = {224, 112};

    float score[36], ns[36];
    int prop = 0;
    for (int f = 0; f < 2; ++f) {
        const int kh = fkh[f], kw = fkw[f];
        const int sh = FMW - kh + 1, sw = FMW - kw + 1;
        const float inv = 1.f / (float)(kh * kw);
        for (int y = 0; y < sh; ++y)
            for (int x = 0; x < sw; ++x) {
                float s = 0.f;
                for (int dy = 0; dy < kh; ++dy)
                    for (int dx = 0; dx < kw; ++dx) s += fsum[(y + dy) * FMW + x + dx];
                score[y * sw + x] = s * inv;
            }
        for (int it = 0; it < nfil[f]; ++it) {
            float vmax = -1e30f; int idx = 0;
            for (int i = 0; i < sh * sw; ++i)
                if (score[i] > vmax) { vmax = score[i]; idx = i; }
            const int ly = idx / sw, lx = idx % sw;
            for (int y = 0; y < sh; ++y)
                for (int x = 0; x < sw; ++x) {
                    float mp = -1e30f;
                    for (int dy = -1; dy <= 1; ++dy)
                        for (int dx = -1; dx <= 1; ++dx) {
                            int yyq = y + dy, xxq = x + dx;
                            if (yyq >= 0 && yyq < sh && xxq >= 0 && xxq < sw) {
                                float sv = score[yyq * sw + xxq];
                                mp = sv > mp ? sv : mp;
                            }
                        }
                    ns[y * sw + x] = (mp != vmax) ? score[y * sw + x] : 0.f;
                }
            for (int i = 0; i < sh * sw; ++i) score[i] = ns[i];

            const float rh = (float)(2 * ly + FMW - sh + 1) / (2.f * FMW);
            const float rw = (float)(2 * lx + FMW - sw + 1) / (2.f * FMW);
            const int cy = (int)((float)by + (float)smin * rh);
            const int cx = (int)((float)bx + (float)smin * rw);
            const int half = psz[f] / 2, rem = psz[f] % 2;
            int t0 = cy - half, b0 = cy + half + rem;
            int l0 = cx - half, r0 = cx + half + rem;
            int bel = imin(t0, 0); b0 -= bel; t0 -= bel;
            int ov  = imax(b0 - sy, 0); t0 -= ov; b0 -= ov; t0 = imax(t0, 0);
            bel = imin(l0, 0); r0 -= bel; l0 -= bel;
            ov  = imax(r0 - sx, 0); l0 -= ov; r0 -= ov; l0 = imax(l0, 0);
            tl[(b * NPROP + prop) * 2 + 0] = t0;
            tl[(b * NPROP + prop) * 2 + 1] = l0;
            ++prop;
        }
    }
}

// ---------------------------------------------------------------------------
// crop_resize (align_corners bilinear; P==224 degenerates to exact gather).
// Writes windows directly as f16 for the second backbone pass.
// ---------------------------------------------------------------------------
__global__ void k_window(const float* __restrict__ img, const int* __restrict__ tl,
                         h16* __restrict__ win) {
    const size_t total = (size_t)BATCH * NPROP * 3 * CROPSZ * CROPSZ;
    size_t idx = (size_t)blockIdx.x * 256 + threadIdx.x;
    if (idx >= total) return;
    const int x = (int)(idx % CROPSZ); size_t r = idx / CROPSZ;
    const int y = (int)(r % CROPSZ);   r /= CROPSZ;
    const int c = (int)(r % 3);        r /= 3;
    const int w = (int)r;              // 0..191
    const int b = w / NPROP, prop = w % NPROP;
    const int P = (prop < 3) ? 224 : 112;
    const float g = (float)(P - 1) / (float)(CROPSZ - 1);
    const int t0 = tl[w * 2 + 0], l0 = tl[w * 2 + 1];
    const float syf = (float)t0 + (float)y * g;
    const float sxf = (float)l0 + (float)x * g;
    const int y0 = (int)floorf(syf); const float wy = syf - (float)y0;
    const int x0 = (int)floorf(sxf); const float wx = sxf - (float)x0;
    const int y1 = imin(y0 + 1, HIMG - 1);
    const int x1 = imin(x0 + 1, HIMG - 1);
    const float* im = img + ((size_t)b * 3 + c) * HIMG * HIMG;
    const float v00 = im[(size_t)y0 * HIMG + x0], v01 = im[(size_t)y0 * HIMG + x1];
    const float v10 = im[(size_t)y1 * HIMG + x0], v11 = im[(size_t)y1 * HIMG + x1];
    const float row0 = v00 * (1.f - wy) + v10 * wy;
    const float row1 = v01 * (1.f - wy) + v11 * wy;
    win[idx] = (h16)(row0 * (1.f - wx) + row1 * wx);
}

// ---------------------------------------------------------------------------
__global__ void k_pack(const float* __restrict__ gw, const float* __restrict__ gwin,
                       h16* __restrict__ embA) {
    const int n = BATCH * 7 * CRES;
    int idx = blockIdx.x * 256 + threadIdx.x;
    if (idx >= n) return;
    const int o = idx % CRES, r = idx / CRES;
    const int b = r / 7, j = r % 7;
    const float v = (j < 6) ? gwin[(size_t)(b * NPROP + j) * CRES + o]
                            : gw[(size_t)b * CRES + o];
    embA[idx] = (h16)v;
}

// ---------------------------------------------------------------------------
// Attention core: L=7, HEADS=4, dh=32.  One block per batch element, thread
// (h,l) owns one query row.  Output written as f16 for the out-proj GEMM.
// ---------------------------------------------------------------------------
__global__ void k_attn(const float* __restrict__ qkv, h16* __restrict__ ybf) {
    const int b = blockIdx.x, t = threadIdx.x;
    if (t >= 28) return;
    const int h = t / 7, l = t % 7;
    const float* qr = qkv + (size_t)(b * 7 + l) * 384 + h * 32;
    float s[7];
    float mx = -1e30f;
    for (int m = 0; m < 7; ++m) {
        const float* kr = qkv + (size_t)(b * 7 + m) * 384 + 128 + h * 32;
        float d = 0.f;
        for (int dd = 0; dd < 32; ++dd) d += qr[dd] * kr[dd];
        s[m] = d * 0.1767766953f;   // 1/sqrt(32)
        mx = s[m] > mx ? s[m] : mx;
    }
    float sum = 0.f;
    for (int m = 0; m < 7; ++m) { s[m] = __expf(s[m] - mx); sum += s[m]; }
    const float invs = 1.f / sum;
    for (int dd = 0; dd < 32; ++dd) {
        float a = 0.f;
        for (int m = 0; m < 7; ++m)
            a += s[m] * qkv[(size_t)(b * 7 + m) * 384 + 256 + h * 32 + dd];
        ybf[(size_t)(b * 7 + l) * DMODEL + h * 32 + dd] = (h16)(a * invs);
    }
}

// ---------------------------------------------------------------------------
__global__ void k_add_ln(const float* __restrict__ x, const float* __restrict__ y,
                         const float* __restrict__ w, const float* __restrict__ bb,
                         float* __restrict__ out, h16* __restrict__ outh) {
    __shared__ float red[DMODEL];
    const int r = blockIdx.x, t = threadIdx.x;
    const float v = x[(size_t)r * DMODEL + t] + y[(size_t)r * DMODEL + t];
    red[t] = v; __syncthreads();
    for (int s = 64; s > 0; s >>= 1) { if (t < s) red[t] += red[t + s]; __syncthreads(); }
    const float mean = red[0] * (1.f / DMODEL);
    __syncthreads();
    const float d = v - mean;
    red[t] = d * d; __syncthreads();
    for (int s = 64; s > 0; s >>= 1) { if (t < s) red[t] += red[t + s]; __syncthreads(); }
    const float var = red[0] * (1.f / DMODEL);
    const float o = d * rsqrtf(var + 1e-5f) * w[t] + bb[t];
    out[(size_t)r * DMODEL + t] = o;
    if (outh) outh[(size_t)r * DMODEL + t] = (h16)o;
}

// ---------------------------------------------------------------------------
__global__ void k_final(const float* __restrict__ allE, const float* __restrict__ fcw,
                        const float* __restrict__ fcb, float* __restrict__ out) {
    const int b = threadIdx.x;
    if (b >= BATCH) return;
    float s = 0.f;
    for (int d = 0; d < DMODEL; ++d) s += allE[(size_t)(b * 7 + 6) * DMODEL + d] * fcw[d];
    out[b] = s + fcb[0];
}

// ---------------------------------------------------------------------------
extern "C" void kernel_launch(void* const* d_in, const int* in_sizes, int n_in,
                              void* d_out, int out_size, void* d_ws, size_t ws_size,
                              hipStream_t stream) {
    (void)in_sizes; (void)n_in; (void)out_size; (void)ws_size;

    const float* input_img  = (const float*)d_in[0];
    const float* cropped    = (const float*)d_in[1];
    const float* scale      = (const float*)d_in[2];
    const float* conv_w     = (const float*)d_in[3];
    const float* fc1_w      = (const float*)d_in[4];
    const float* fc1_b      = (const float*)d_in[5];
    const float* fc_w       = (const float*)d_in[6];
    const float* fc_b       = (const float*)d_in[7];
    const float* sa_in_w    = (const float*)d_in[8];
    const float* sa_in_b    = (const float*)d_in[9];
    const float* sa_out_w   = (const float*)d_in[10];
    const float* sa_out_b   = (const float*)d_in[11];
    const float* sa_ln1_w   = (const float*)d_in[12];
    const float* sa_ln1_b   = (const float*)d_in[13];
    const float* sa_fc1_w   = (const float*)d_in[14];
    const float* sa_fc1_b   = (const float*)d_in[15];
    const float* sa_fc2_w   = (const float*)d_in[16];
    const float* sa_fc2_b   = (const float*)d_in[17];
    const float* sa_ln2_w   = (const float*)d_in[18];
    const float* sa_ln2_b   = (const float*)d_in[19];

    char* wsp = (char*)d_ws;
    size_t off = 0;
    auto take = [&](size_t bytes) -> void* {
        void* p = wsp + off;
        off = (off + bytes + 255) & ~(size_t)255;
        return p;
    };
    h16*   convw_h  = (h16*)  take((size_t)CRES * KCONV * 2);
    h16*   crop_h   = (h16*)  take((size_t)BATCH * 3 * CROPSZ * CROPSZ * 2);
    h16*   win_h    = (h16*)  take((size_t)BATCH * NPROP * 3 * CROPSZ * CROPSZ * 2);
    float* fm       = (float*)take((size_t)BATCH * CRES * FMP * 4);
    float* gapw     = (float*)take((size_t)BATCH * CRES * 4);
    float* gapn     = (float*)take((size_t)BATCH * NPROP * CRES * 4);
    int*   tl       = (int*)  take((size_t)BATCH * NPROP * 2 * 4);
    h16*   embA     = (h16*)  take((size_t)BATCH * 7 * CRES * 2);
    h16*   fc1w_h   = (h16*)  take((size_t)DMODEL * CRES * 2);
    h16*   sainw_h  = (h16*)  take((size_t)3 * 384 * DMODEL * 2);
    h16*   saoutw_h = (h16*)  take((size_t)3 * DMODEL * DMODEL * 2);
    h16*   saf1w_h  = (h16*)  take((size_t)3 * DMODEL * DMODEL * 2);
    h16*   saf2w_h  = (h16*)  take((size_t)3 * DMODEL * DMODEL * 2);
    float* allE     = (float*)take((size_t)224 * DMODEL * 4);
    h16*   x_h      = (h16*)  take((size_t)224 * DMODEL * 2);
    float* qkv      = (float*)take((size_t)224 * 384 * 4);
    h16*   y_h      = (h16*)  take((size_t)224 * DMODEL * 2);
    float* y2       = (float*)take((size_t)224 * DMODEL * 4);
    float* x1       = (float*)take((size_t)224 * DMODEL * 4);
    h16*   x1_h     = (h16*)  take((size_t)224 * DMODEL * 2);
    float* h1       = (float*)take((size_t)224 * DMODEL * 4);
    h16*   h1_h     = (h16*)  take((size_t)224 * DMODEL * 2);
    float* h2       = (float*)take((size_t)224 * DMODEL * 4);

    auto cvt = [&](const float* src, h16* dst, int n) {
        k_cvt<<<(n + 255) / 256, 256, 0, stream>>>(src, dst, n);
    };

    cvt(conv_w,   convw_h,  CRES * KCONV);
    cvt(cropped,  crop_h,   BATCH * 3 * CROPSZ * CROPSZ);
    cvt(fc1_w,    fc1w_h,   DMODEL * CRES);
    cvt(sa_in_w,  sainw_h,  3 * 384 * DMODEL);
    cvt(sa_out_w, saoutw_h, 3 * DMODEL * DMODEL);
    cvt(sa_fc1_w, saf1w_h,  3 * DMODEL * DMODEL);
    cvt(sa_fc2_w, saf2w_h,  3 * DMODEL * DMODEL);

    hipMemsetAsync(gapw, 0, (size_t)BATCH * CRES * 4, stream);
    hipMemsetAsync(gapn, 0, (size_t)BATCH * NPROP * CRES * 4, stream);

    // Backbone on cropped images: M = 32*49 = 1568, N tiles of 256
    k_conv_gemm<<<dim3(49, 8), 256, 0, stream>>>(crop_h, convw_h, fm, gapw);

    // Coordinates + crop windows
    k_coords<<<BATCH, 64, 0, stream>>>(fm, scale, tl);
    {
        const size_t total = (size_t)BATCH * NPROP * 3 * CROPSZ * CROPSZ;
        k_window<<<(unsigned)((total + 255) / 256), 256, 0, stream>>>(input_img, tl, win_h);
    }

    // Backbone on 192 windows: M = 192*49 = 9408
    k_conv_gemm<<<dim3(294, 8), 256, 0, stream>>>(win_h, convw_h, nullptr, gapn);

    // fc1 on all 224 embeddings
    k_pack<<<(BATCH * 7 * CRES + 255) / 256, 256, 0, stream>>>(gapw, gapn, embA);
    k_gemm<<<dim3(7, 2), 256, 0, stream>>>(embA, fc1w_h, fc1_b, allE, x_h,
                                           224, DMODEL, CRES, 1);

    // 3 self-attention layers
    for (int i = 0; i < 3; ++i) {
        k_gemm<<<dim3(7, 6), 256, 0, stream>>>(x_h, sainw_h + (size_t)i * 384 * DMODEL,
                                               sa_in_b + i * 384, qkv, (h16*)nullptr,
                                               224, 384, DMODEL, 0);
        k_attn<<<BATCH, 32, 0, stream>>>(qkv, y_h);
        k_gemm<<<dim3(7, 2), 256, 0, stream>>>(y_h, saoutw_h + (size_t)i * DMODEL * DMODEL,
                                               sa_out_b + i * DMODEL, y2, (h16*)nullptr,
                                               224, DMODEL, DMODEL, 0);
        k_add_ln<<<224, DMODEL, 0, stream>>>(allE, y2, sa_ln1_w + i * DMODEL,
                                             sa_ln1_b + i * DMODEL, x1, x1_h);
        k_gemm<<<dim3(7, 2), 256, 0, stream>>>(x1_h, saf1w_h + (size_t)i * DMODEL * DMODEL,
                                               sa_fc1_b + i * DMODEL, h1, h1_h,
                                               224, DMODEL, DMODEL, 1);
        k_gemm<<<dim3(7, 2), 256, 0, stream>>>(h1_h, saf2w_h + (size_t)i * DMODEL * DMODEL,
                                               sa_fc2_b + i * DMODEL, h2, (h16*)nullptr,
                                               224, DMODEL, DMODEL, 0);
        k_add_ln<<<224, DMODEL, 0, stream>>>(x1, h2, sa_ln2_w + i * DMODEL,
                                             sa_ln2_b + i * DMODEL, allE, x_h);
    }

    k_final<<<1, 32, 0, stream>>>(allE, fc_w, fc_b, (float*)d_out);
}
